// MultiHeadAttention_2800318677657
// MI455X (gfx1250) — matrix operand order repaired and checked by reading, not compiled, not run on hardware
//
#include <hip/hip_runtime.h>
#include <hip/hip_bf16.h>

typedef __bf16 bf16;
typedef __attribute__((ext_vector_type(16))) __bf16 v16bf;
typedef __attribute__((ext_vector_type(8)))  __bf16 v8bf;
typedef __attribute__((ext_vector_type(8)))  float  v8f;
typedef __attribute__((ext_vector_type(4)))  float  v4f;

#define DMODEL 1024
#define NHEADS 16
#define DK     64
#define SEQ    2048
#define BATCH  2
#define NEGINF -1000000000.0f

// gfx1250 async global->LDS staging (ASYNCcnt path). Set to 0 to fall back to
// plain LDS stores if the asm is rejected.
#define USE_ASYNC_STAGING 1

union FragAB { v16bf v; v8bf h[2]; bf16 e[16]; };
union FragC  { v8f   v; float e[8]; };

// ---- WMMA wrapper: D = A(16x32 bf16) * B(32x16 bf16) + C(16x16 f32) ----
__device__ __forceinline__ v8f wmma_bf16(v16bf a, v16bf b, v8f c) {
    return __builtin_amdgcn_wmma_f32_16x16x32_bf16(
        /*neg_a=*/false, a, /*neg_b=*/false, b,
        /*c_mod=*/(short)0, c, /*reuse_a=*/false, /*reuse_b=*/false);
}

// A fragment: 16x32 (MxK) bf16. base -> element (m0,k0) of row-major [M][ld].
// ISA layout: lane lo (0..15) = M row; elems 0..7 = K[8*hi..+8), elems 8..15 = K[16+8*hi..+8)
__device__ __forceinline__ v16bf load_frag_a(const bf16* base, int ld, int lane) {
    const int lo = lane & 15, hi = lane >> 4;
    const bf16* row = base + lo * ld + 8 * hi;
    FragAB f;
    f.h[0] = *(const v8bf*)(row);
    f.h[1] = *(const v8bf*)(row + 16);
    return f.v;
}

// B fragment: 32x16 (KxN) bf16. base -> element (k0,n0) of row-major [K][ld].
// ISA layout: lane lo (0..15) = N col; elems 0..7 = K[8*hi..+8), elems 8..15 = K[16+8*hi..+8)
__device__ __forceinline__ v16bf load_frag_b(const bf16* base, int ld, int lane) {
    const int lo = lane & 15, hi = lane >> 4;
    FragAB f;
    #pragma unroll
    for (int s = 0; s < 8; ++s) {
        f.e[s]     = base[(8 * hi + s) * ld + lo];
        f.e[8 + s] = base[(16 + 8 * hi + s) * ld + lo];
    }
    return f.v;
}

#if USE_ASYNC_STAGING
// One 16-byte async DMA chunk: LDS[lds_generic_ptr] = MEM[gsrc .. +16)
// ISA 10.2: LDS aperture generic address maps to LDS by truncation to [31:0].
__device__ __forceinline__ void async_copy_b128(void* lds_ptr, const void* gsrc) {
    unsigned lds_off = (unsigned)(size_t)lds_ptr;
    asm volatile("global_load_async_to_lds_b128 %0, %1, off"
                 :: "v"(lds_off), "v"(gsrc) : "memory");
}
__device__ __forceinline__ void async_wait0() {
    asm volatile("s_wait_asynccnt 0x0" ::: "memory");
}
#endif

// ---------------------------------------------------------------------------
// Kernel 1: W (fp32, [N][K]) -> Wt (bf16, [K][N])   (einsum 'bsd,ed->bse' => y = x * W^T)
// ---------------------------------------------------------------------------
__global__ void __launch_bounds__(256)
transpose_convert_kernel(const float* __restrict__ w, bf16* __restrict__ wt) {
    __shared__ float tile[64][65];
    const int n0 = blockIdx.x * 64, k0 = blockIdx.y * 64;
    const int t = threadIdx.x;
    #pragma unroll
    for (int i = 0; i < 16; ++i) {
        int r = i * 4 + (t >> 6), c = t & 63;
        tile[r][c] = w[(size_t)(n0 + r) * DMODEL + (k0 + c)];
    }
    __syncthreads();
    #pragma unroll
    for (int i = 0; i < 16; ++i) {
        int r = i * 4 + (t >> 6), c = t & 63;
        wt[(size_t)(k0 + r) * DMODEL + (n0 + c)] = (bf16)tile[c][r];
    }
}

// ---------------------------------------------------------------------------
// Kernel 2: GEMM  C[m,n] = sum_k A[m,k]*Bt[k,n] + bias[n]
//   A: [4096][1024] fp32 (A_BF16=0) or bf16 (A_BF16=1), Bt: [1024][1024] bf16
//   OUT_MODE 0: fp32 row-major [m][n]            (final output projection)
//   OUT_MODE 1: bf16 [b][h][s][d]                (Q, V)
//   OUT_MODE 2: bf16 [b][h][d][s]  (transposed)  (K)
// Block: 256 thr (8 waves), tile 64(M) x 64(N), K step 32.
// B tile staged via async global->LDS DMA; A tile converted fp32->bf16 in flight.
// ---------------------------------------------------------------------------
template<bool A_BF16, int OUT_MODE>
__global__ void __launch_bounds__(256)
gemm_kernel(const void* __restrict__ Aptr, const bf16* __restrict__ Bt,
            const float* __restrict__ bias, void* __restrict__ out) {
    __shared__ bf16 As[64][40];   // 64x32 tile (+pad, 80B row stride: 16B aligned)
    __shared__ bf16 Bs[32][72];   // 32x64 tile (+pad, 144B row stride: 16B aligned)
    const int t = threadIdx.x, wave = t >> 5, lane = t & 31;
    const int lo = lane & 15, hi = lane >> 4;
    const int m0 = blockIdx.x * 64;
    const int n0 = blockIdx.y * 64;
    const int mi = wave & 3;              // M subtile 0..3
    const int nb = (wave >> 2) * 2;       // N subtiles nb, nb+1
    const float* Af = (const float*)Aptr;
    const bf16*  Ab = (const bf16*)Aptr;

    v8f c0 = {}, c1 = {};
    for (int k0 = 0; k0 < DMODEL; k0 += 32) {
        // ---- stage B 32x64 bf16 (pure copy): one 16B chunk per thread
        {
            const int r = t >> 3, cb = (t & 7) * 16;   // row, byte col (0..112)
            const char* gsrc = (const char*)&Bt[(size_t)(k0 + r) * DMODEL + n0] + cb;
#if USE_ASYNC_STAGING
            async_copy_b128((char*)&Bs[r][0] + cb, gsrc);
#else
            *(v8bf*)((char*)&Bs[r][0] + cb) = *(const v8bf*)gsrc;
#endif
        }
        // ---- stage A 64x32 -> bf16: 8 consecutive elements per thread
        {
            const int r = t >> 2, c = (t & 3) * 8;
            if (A_BF16) {
                *(v8bf*)&As[r][c] = *(const v8bf*)(Ab + (size_t)(m0 + r) * DMODEL + k0 + c);
            } else {
                const float* src = Af + (size_t)(m0 + r) * DMODEL + k0 + c;
                v4f a0 = *(const v4f*)src, a1 = *(const v4f*)(src + 4);
                v8bf x;
                #pragma unroll
                for (int i = 0; i < 4; ++i) { x[i] = (bf16)a0[i]; x[4 + i] = (bf16)a1[i]; }
                *(v8bf*)&As[r][c] = x;
            }
        }
        if (k0 + 32 < DMODEL)
            __builtin_prefetch(&Bt[(size_t)(k0 + 32) * DMODEL + n0], 0, 0);
#if USE_ASYNC_STAGING
        async_wait0();
#endif
        __syncthreads();
        v16bf a  = load_frag_a(&As[mi * 16][0], 40, lane);
        v16bf b0 = load_frag_b(&Bs[0][nb * 16],      72, lane);
        v16bf b1 = load_frag_b(&Bs[0][nb * 16 + 16], 72, lane);
        c0 = wmma_bf16(a, b0, c0);
        c1 = wmma_bf16(a, b1, c1);
        __syncthreads();
    }
    // epilogue: C elem j -> (M = j + 8*hi, N = lo)
    #pragma unroll
    for (int f = 0; f < 2; ++f) {
        FragC u; u.v = f ? c1 : c0;
        const int gn = n0 + (nb + f) * 16 + lo;
        const float bv = bias[gn];
        #pragma unroll
        for (int j = 0; j < 8; ++j) {
            const int gm = m0 + mi * 16 + j + 8 * hi;
            const float val = u.e[j] + bv;
            if (OUT_MODE == 0) {
                ((float*)out)[(size_t)gm * DMODEL + gn] = val;
            } else {
                const int b = gm / SEQ, s = gm % SEQ;
                const int h = gn / DK,  d = gn % DK;
                if (OUT_MODE == 1)
                    ((bf16*)out)[(((size_t)(b * NHEADS + h) * SEQ) + s) * DK + d] = (bf16)val;
                else
                    ((bf16*)out)[(((size_t)(b * NHEADS + h) * DK) + d) * SEQ + s] = (bf16)val;
            }
        }
    }
}

// ---------------------------------------------------------------------------
// Kernel 3: fused attention. One workgroup = one (b,h, 32-row q tile).
// 32x2048 fp32 score strip lives in 256KB dynamic LDS (CDNA5: 320KB/WGP).
//  phase 1: scores = Q K^T / 8, causal mask -> LDS        (WMMA bf16)
//  phase 2: softmax in LDS, stream weights to d_out (non-temporal: written once,
//           keeps the 192MB L2 free for K/V reuse across workgroups)
//  phase 3: attn_out = P V                                 (WMMA bf16)
// ---------------------------------------------------------------------------
__global__ void __launch_bounds__(256)
attn_kernel(const bf16* __restrict__ Q,   // [B*H][S][64]
            const bf16* __restrict__ Kt,  // [B*H][64][S]
            const bf16* __restrict__ V,   // [B*H][S][64]
            float* __restrict__ wout,     // [B*H][S][S]
            bf16* __restrict__ aout) {    // [B][S][1024]
    extern __shared__ float sc[];         // [32][SEQ]
    const int t = threadIdx.x, wave = t >> 5, lane = t & 31;
    const int lo = lane & 15, hi = lane >> 4;
    const int bh = blockIdx.x >> 6;               // 64 q-tiles per (b,h)
    const int q0 = (blockIdx.x & 63) * 32;
    const int b = bh >> 4, h = bh & 15;
    const bf16* Qb  = Q  + (size_t)bh * SEQ * DK;
    const bf16* Ktb = Kt + (size_t)bh * DK * SEQ;
    const bf16* Vb  = V  + (size_t)bh * SEQ * DK;
    const int qhalf = wave >> 2;                  // rows [qhalf*16, +16)
    const int m0 = q0 + qhalf * 16;
    const int ktmax = (q0 >> 4) + 2;              // #16-wide k-tiles touching mask
    const int kpad  = ktmax * 16;                 // == q0 + 32

    // ---- phase 1: scores -> LDS
    v16bf qa0 = load_frag_a(Qb + (size_t)m0 * DK,      DK, lane);
    v16bf qa1 = load_frag_a(Qb + (size_t)m0 * DK + 32, DK, lane);
    for (int kt = (wave & 3); kt < ktmax; kt += 4) {
        const int kk0 = kt * 16;
        v16bf kb0 = load_frag_b(Ktb + kk0,                    SEQ, lane); // d = 0..31
        v16bf kb1 = load_frag_b(Ktb + (size_t)32 * SEQ + kk0, SEQ, lane); // d = 32..63
        v8f c = {};
        c = wmma_bf16(qa0, kb0, c);
        c = wmma_bf16(qa1, kb1, c);
        FragC u; u.v = c;
        #pragma unroll
        for (int j = 0; j < 8; ++j) {
            const int ql = qhalf * 16 + j + 8 * hi;   // local row 0..31
            const int kg = kk0 + lo;
            float s = u.e[j] * 0.125f;                 // 1/sqrt(64)
            if (kg > q0 + ql) s = NEGINF;              // causal mask
            sc[ql * SEQ + kg] = s;
        }
    }
    __syncthreads();

    // ---- phase 2: softmax + write attn_weights (non-temporal stream)
    for (int r = wave; r < 32; r += 8) {
        const int qg = q0 + r;
        float* row = sc + r * SEQ;
        float mx = NEGINF;
        for (int k = lane; k <= qg; k += 32) mx = fmaxf(mx, row[k]);
        #pragma unroll
        for (int off = 16; off > 0; off >>= 1) mx = fmaxf(mx, __shfl_xor(mx, off, 32));
        float sum = 0.f;
        for (int k = lane; k <= qg; k += 32) {
            float e = __expf(row[k] - mx);
            row[k] = e;
            sum += e;
        }
        #pragma unroll
        for (int off = 16; off > 0; off >>= 1) sum += __shfl_xor(sum, off, 32);
        const float inv = 1.0f / sum;
        float* wrow = wout + ((size_t)bh * SEQ + qg) * SEQ;
        for (int k = lane; k < SEQ; k += 32) {
            float p = 0.f;
            if (k <= qg) { p = row[k] * inv; row[k] = p; }
            else if (k < kpad) row[k] = 0.f;           // zero pad for P*V tiles
            __builtin_nontemporal_store(p, wrow + k);  // written once, never re-read
        }
    }
    __syncthreads();

    // ---- phase 3: attn_out = P(32 x kpad) * V(kpad x 64)
    const int d0 = (wave & 3) * 16;
    v8f acc = {};
    for (int kk0 = 0; kk0 < kpad; kk0 += 32) {
        FragAB pa;                                     // P fragment, fp32->bf16
        const float* prow = sc + (qhalf * 16 + lo) * SEQ + kk0;
        v4f f0 = *(const v4f*)(prow + 8 * hi);
        v4f f1 = *(const v4f*)(prow + 8 * hi + 4);
        v4f f2 = *(const v4f*)(prow + 16 + 8 * hi);
        v4f f3 = *(const v4f*)(prow + 16 + 8 * hi + 4);
        #pragma unroll
        for (int i = 0; i < 4; ++i) {
            pa.e[i]      = (bf16)f0[i];
            pa.e[4 + i]  = (bf16)f1[i];
            pa.e[8 + i]  = (bf16)f2[i];
            pa.e[12 + i] = (bf16)f3[i];
        }
        v16bf vb = load_frag_b(Vb + (size_t)kk0 * DK + d0, DK, lane);
        acc = wmma_bf16(pa.v, vb, acc);
    }
    FragC u; u.v = acc;
    #pragma unroll
    for (int j = 0; j < 8; ++j) {
        const int qg = q0 + qhalf * 16 + j + 8 * hi;
        aout[((size_t)b * SEQ + qg) * DMODEL + h * DK + d0 + lo] = (bf16)u.e[j];
    }
}

// ---------------------------------------------------------------------------
extern "C" void kernel_launch(void* const* d_in, const int* in_sizes, int n_in,
                              void* d_out, int out_size, void* d_ws, size_t ws_size,
                              hipStream_t stream) {
    const float* query = (const float*)d_in[0];
    const float* key   = (const float*)d_in[1];
    const float* value = (const float*)d_in[2];
    // d_in[3] = mask (int32 tril) -- causal structure applied analytically
    const float* wq = (const float*)d_in[4];
    const float* bq = (const float*)d_in[5];
    const float* wk = (const float*)d_in[6];
    const float* bk = (const float*)d_in[7];
    const float* wv = (const float*)d_in[8];
    const float* bv = (const float*)d_in[9];
    const float* wo = (const float*)d_in[10];
    const float* bo = (const float*)d_in[11];

    char* ws = (char*)d_ws;                    // 40 MB used
    bf16* wt_q = (bf16*)(ws + (size_t)0);
    bf16* wt_k = (bf16*)(ws + ((size_t)2  << 20));
    bf16* wt_v = (bf16*)(ws + ((size_t)4  << 20));
    bf16* wt_o = (bf16*)(ws + ((size_t)6  << 20));
    bf16* Qbf  = (bf16*)(ws + ((size_t)8  << 20));   // [B*H][S][64]
    bf16* Ktbf = (bf16*)(ws + ((size_t)16 << 20));   // [B*H][64][S]
    bf16* Vbf  = (bf16*)(ws + ((size_t)24 << 20));   // [B*H][S][64]
    bf16* Aout = (bf16*)(ws + ((size_t)32 << 20));   // [B][S][1024]

    float* out_proj = (float*)d_out;                                  // (B,S,D)
    float* out_w    = out_proj + (size_t)BATCH * SEQ * DMODEL;        // (B,H,S,S)

    const dim3 blk(256);

    // 1) weights -> bf16 transposed [K][N]
    const dim3 tg(DMODEL / 64, DMODEL / 64);
    transpose_convert_kernel<<<tg, blk, 0, stream>>>(wq, wt_q);
    transpose_convert_kernel<<<tg, blk, 0, stream>>>(wk, wt_k);
    transpose_convert_kernel<<<tg, blk, 0, stream>>>(wv, wt_v);
    transpose_convert_kernel<<<tg, blk, 0, stream>>>(wo, wt_o);

    // 2) projections (M=4096, N=1024, K=1024)
    const dim3 gg(BATCH * SEQ / 64, DMODEL / 64);
    gemm_kernel<false, 1><<<gg, blk, 0, stream>>>((const void*)query, wt_q, bq, (void*)Qbf);
    gemm_kernel<false, 2><<<gg, blk, 0, stream>>>((const void*)key,   wt_k, bk, (void*)Ktbf);
    gemm_kernel<false, 1><<<gg, blk, 0, stream>>>((const void*)value, wt_v, bv, (void*)Vbf);

    // 3) fused attention: scores + mask + softmax + weights-out + P*V
    const int attn_blocks = BATCH * NHEADS * (SEQ / 32);      // 2048
    const size_t smem = (size_t)32 * SEQ * sizeof(float);     // 256 KB LDS strip
    attn_kernel<<<attn_blocks, blk, smem, stream>>>(Qbf, Ktbf, Vbf, out_w, Aout);

    // 4) output projection -> fp32 d_out
    gemm_kernel<true, 0><<<gg, blk, 0, stream>>>((const void*)Aout, wt_o, bo, (void*)out_proj);
}
